// GAT_48919677501666
// MI455X (gfx1250) — matrix operand order repaired and checked by reading, not compiled, not run on hardware
//
#include <hip/hip_runtime.h>
#include <hip/hip_bf16.h>

typedef __attribute__((ext_vector_type(16))) _Float16 v16h;
typedef __attribute__((ext_vector_type(8)))  float    v8f;

#define NNODES 100000
#define HIDDIM 64
#define NGRAPH 64
#define NEG_SLOPE 0.2f
#define BN_EPS 1e-5f
#define NEG_FLT_MAX (-3.402823466e38f)

// ---------------------------------------------------------------- utilities
// vectorized fill: n4 = count of float4 elements
__global__ void fill4_kernel(float4* __restrict__ p, long long n4, float v) {
    long long i = (long long)blockIdx.x * blockDim.x + threadIdx.x;
    long long stride = (long long)gridDim.x * blockDim.x;
    float4 val = make_float4(v, v, v, v);
    for (; i < n4; i += stride) p[i] = val;
}

__device__ __forceinline__ void atomicMaxFloat(float* addr, float value) {
    if (value >= 0.0f) {
        atomicMax((int*)addr, __float_as_int(value));
    } else {
        atomicMin((unsigned int*)addr, __float_as_uint(value));
    }
}

// ------------------------------------------------- WMMA GEMM: H = X * W
// X: [M x K] f32 row-major, W: [K x 64] f32 row-major, H: [M x 64] f32.
// W is staged once per block into LDS, pre-converted to f16 and pre-swizzled
// into B-fragment lane order; waves read fragments with ds_load_b128.
// One wave computes a 16(M) x 64(N) strip via v_wmma_f32_16x16x32_f16.
__global__ void wmma_gemm_kernel(const float* __restrict__ X,
                                 const float* __restrict__ Wm,
                                 float* __restrict__ H,
                                 int M, int K) {
    // ---- cooperative W staging: fragment layout [chunk][nt][lane][16 halves]
    __shared__ _Float16 smem[8192];              // up to K=128: 4*4*32*16 = 16KB
    const int chunks = K >> 5;                   // K/32
    const int totalHalves = chunks << 11;        // chunks*4*32*16
    for (int hidx = threadIdx.x; hidx < totalHalves; hidx += blockDim.x) {
        const int idx   = hidx & 15;
        const int slot  = hidx >> 4;
        const int ln    = slot & 31;
        const int nt    = (slot >> 5) & 3;
        const int chunk = slot >> 7;
        const int kk    = (chunk << 5) + ((ln >= 16) ? 8 : 0) + (idx & 7) + ((idx & 8) ? 16 : 0);
        const int col   = (ln & 15) + nt * 16;
        smem[hidx] = (_Float16)Wm[(size_t)kk * HIDDIM + col];
    }
    __syncthreads();

    const int lane = threadIdx.x & 31;
    const int waveInBlock = threadIdx.x >> 5;
    const int mtile = blockIdx.x * (blockDim.x >> 5) + waveInBlock;
    const int mtiles = M >> 4;
    if (mtile >= mtiles) return;        // uniform per-wave: EXEC stays all-1s

    const int  row16  = lane & 15;
    const bool hi     = lane >= 16;
    const int  kbaseA = hi ? 8 : 0;     // A frag: lanes 16-31 carry K+8 group

    v8f acc[4] = {};                    // 4 N-tiles of 16 -> HID=64
    const float* xrow = X + (size_t)(mtile * 16 + row16) * K;

    for (int k0 = 0; k0 < K; k0 += 32) {
        if (k0 + 32 < K) __builtin_prefetch(xrow + k0 + 32, 0, 1);

        // A fragment: two contiguous 8-float runs -> 4x global_load_b128
        const float4 x0 = *(const float4*)(xrow + k0 + kbaseA);
        const float4 x1 = *(const float4*)(xrow + k0 + kbaseA + 4);
        const float4 x2 = *(const float4*)(xrow + k0 + kbaseA + 16);
        const float4 x3 = *(const float4*)(xrow + k0 + kbaseA + 20);
        v16h a;
        a[0]  = (_Float16)x0.x;  a[1]  = (_Float16)x0.y;
        a[2]  = (_Float16)x0.z;  a[3]  = (_Float16)x0.w;
        a[4]  = (_Float16)x1.x;  a[5]  = (_Float16)x1.y;
        a[6]  = (_Float16)x1.z;  a[7]  = (_Float16)x1.w;
        a[8]  = (_Float16)x2.x;  a[9]  = (_Float16)x2.y;
        a[10] = (_Float16)x2.z;  a[11] = (_Float16)x2.w;
        a[12] = (_Float16)x3.x;  a[13] = (_Float16)x3.y;
        a[14] = (_Float16)x3.z;  a[15] = (_Float16)x3.w;

        const int chunk = k0 >> 5;
#pragma unroll
        for (int nt = 0; nt < 4; ++nt) {
            const v16h b = *(const v16h*)(smem + ((((chunk << 2) + nt) << 5) + lane) * 16);
            acc[nt] = __builtin_amdgcn_wmma_f32_16x16x32_f16(
                false, a, false, b, (short)0, acc[nt], false, false);
        }
    }

    // C/D layout: VGPR r -> row r (lanes 0-15) / row 8+r (lanes 16-31)
    const int rowoff = hi ? 8 : 0;
#pragma unroll
    for (int nt = 0; nt < 4; ++nt) {
#pragma unroll
        for (int r = 0; r < 8; ++r) {
            H[(size_t)(mtile * 16 + rowoff + r) * HIDDIM + nt * 16 + (lane & 15)]
                = acc[nt][r];
        }
    }
}

// ---------------------------------------------- per-node attention logits
__global__ void alpha_kernel(const float* __restrict__ h,
                             const float* __restrict__ avec_s,
                             const float* __restrict__ avec_d,
                             float* __restrict__ as, float* __restrict__ ad,
                             int n) {
    int i = blockIdx.x * blockDim.x + threadIdx.x;
    if (i >= n) return;
    const float* hr = h + (size_t)i * HIDDIM;
    float s = 0.f, d = 0.f;
#pragma unroll
    for (int f = 0; f < HIDDIM; ++f) {
        float v = hr[f];
        s = fmaf(v, avec_s[f], s);
        d = fmaf(v, avec_d[f], d);
    }
    as[i] = s; ad[i] = d;
}

// ------------------------------------------------------------ edge passes
__device__ __forceinline__ void edge_ids(const long long* __restrict__ ei,
                                         long long E, long long i,
                                         long long& s, long long& d) {
    if (i < E) { s = ei[i]; d = ei[E + i]; }
    else       { s = i - E; d = s; }          // self loops appended
}

__global__ void edge_max_kernel(const long long* __restrict__ ei,
                                long long E, long long Etot,
                                const float* __restrict__ as,
                                const float* __restrict__ ad,
                                float* __restrict__ mbuf) {
    long long i = (long long)blockIdx.x * blockDim.x + threadIdx.x;
    if (i >= Etot) return;
    long long s, d; edge_ids(ei, E, i, s, d);
    float e = as[s] + ad[d];
    e = e > 0.f ? e : NEG_SLOPE * e;
    atomicMaxFloat(mbuf + d, e);
}

__global__ void edge_sum_kernel(const long long* __restrict__ ei,
                                long long E, long long Etot,
                                const float* __restrict__ as,
                                const float* __restrict__ ad,
                                const float* __restrict__ mbuf,
                                float* __restrict__ sbuf) {
    long long i = (long long)blockIdx.x * blockDim.x + threadIdx.x;
    if (i >= Etot) return;
    long long s, d; edge_ids(ei, E, i, s, d);
    float e = as[s] + ad[d];
    e = e > 0.f ? e : NEG_SLOPE * e;
    atomicAdd(sbuf + d, __expf(e - mbuf[d]));
}

// one wave32 per edge, 2 features per lane (HID=64)
__global__ void edge_agg_kernel(const long long* __restrict__ ei,
                                long long E, long long Etot,
                                const float* __restrict__ as,
                                const float* __restrict__ ad,
                                const float* __restrict__ mbuf,
                                const float* __restrict__ sbuf,
                                const float* __restrict__ h,
                                float* __restrict__ outp) {
    long long wid = ((long long)blockIdx.x * blockDim.x + threadIdx.x) >> 5;
    const int lane = threadIdx.x & 31;
    if (wid >= Etot) return;
    long long s, d; edge_ids(ei, E, wid, s, d);
    float e = as[s] + ad[d];
    e = e > 0.f ? e : NEG_SLOPE * e;
    float w = __expf(e - mbuf[d]);
    float alpha = w / (sbuf[d] + 1e-16f);
    const float2 hv = *(const float2*)(h + (size_t)s * HIDDIM + lane * 2);
    float* o = outp + (size_t)d * HIDDIM + lane * 2;
    atomicAdd(o,     hv.x * alpha);
    atomicAdd(o + 1, hv.y * alpha);
}

// --------------------------------------------------------------- batchnorm
// coalesced reads (4 rows per block-pass), LDS partial reduction (ds_add_f32),
// then 128 global atomics per block.
__global__ void bn_stats_kernel(const float* __restrict__ h,
                                float* __restrict__ stats, int n) {
    __shared__ float ls[2 * HIDDIM];
    const int t = threadIdx.x;                 // blockDim == 256
    if (t < 2 * HIDDIM) ls[t] = 0.f;
    __syncthreads();

    const int f      = t & (HIDDIM - 1);
    const int rowoff = t >> 6;                 // 0..3
    float s = 0.f, ss = 0.f;
    for (long long i = (long long)blockIdx.x * 4 + rowoff; i < n;
         i += (long long)gridDim.x * 4) {
        float v = h[i * HIDDIM + f];
        s += v; ss += v * v;
    }
    atomicAdd(&ls[f], s);
    atomicAdd(&ls[HIDDIM + f], ss);
    __syncthreads();
    if (t < 2 * HIDDIM) atomicAdd(&stats[t], ls[t]);
}

__global__ void bn_apply_kernel(const float* __restrict__ in,
                                float* __restrict__ outp,
                                const float* __restrict__ stats,
                                const float* __restrict__ gamma,
                                const float* __restrict__ beta,
                                int n, int relu) {
    long long idx = (long long)blockIdx.x * blockDim.x + threadIdx.x;
    if (idx >= (long long)n * HIDDIM) return;
    int f = (int)(idx & (HIDDIM - 1));
    float inv_n = 1.0f / (float)n;
    float mean = stats[f] * inv_n;
    float var  = stats[HIDDIM + f] * inv_n - mean * mean;
    float v = (in[idx] - mean) * rsqrtf(var + BN_EPS) * gamma[f] + beta[f];
    if (relu) v = fmaxf(v, 0.f);
    outp[idx] = v;
}

// -------------------------------------------------------- pooling + final
__global__ void pool_kernel(const float* __restrict__ h,
                            const long long* __restrict__ batch,
                            float* __restrict__ pooled,
                            float* __restrict__ counts, int n) {
    long long idx = (long long)blockIdx.x * blockDim.x + threadIdx.x;
    if (idx >= (long long)n * HIDDIM) return;
    int node = (int)(idx >> 6);
    int f = (int)(idx & 63);
    int g = (int)batch[node];
    atomicAdd(&pooled[(size_t)g * HIDDIM + f], h[idx]);
    if (f == 0) atomicAdd(&counts[g], 1.0f);
}

__global__ void final_kernel(const float* __restrict__ pooled,
                             const float* __restrict__ counts,
                             const float* __restrict__ fcW,
                             const float* __restrict__ fcb,
                             float* __restrict__ outp) {
    int g = threadIdx.x;
    if (g >= NGRAPH) return;
    float c = fmaxf(counts[g], 1.0f);
    float acc = 0.f;
#pragma unroll
    for (int f = 0; f < HIDDIM; ++f)
        acc = fmaf(pooled[(size_t)g * HIDDIM + f], fcW[f], acc);
    outp[g] = acc / c + fcb[0];
}

// ------------------------------------------------------------------ driver
extern "C" void kernel_launch(void* const* d_in, const int* in_sizes, int n_in,
                              void* d_out, int out_size, void* d_ws, size_t ws_size,
                              hipStream_t stream) {
    const float*     x     = (const float*)d_in[0];
    const long long* ei    = (const long long*)d_in[1];
    const long long* batch = (const long long*)d_in[2];
    const float* W1   = (const float*)d_in[3];
    const float* av1s = (const float*)d_in[4];
    const float* av1d = (const float*)d_in[5];
    // d_in[6] = b1: bias before BatchNorm is a no-op (BN is shift-invariant)
    const float* g1   = (const float*)d_in[7];
    const float* be1  = (const float*)d_in[8];
    const float* W2   = (const float*)d_in[9];
    const float* av2s = (const float*)d_in[10];
    const float* av2d = (const float*)d_in[11];
    // d_in[12] = b2: same, folded out by BN
    const float* g2   = (const float*)d_in[13];
    const float* be2  = (const float*)d_in[14];
    const float* fcW  = (const float*)d_in[15];
    const float* fcb  = (const float*)d_in[16];
    float* outp = (float*)d_out;

    const long long E    = (long long)in_sizes[1] / 2;
    const long long Etot = E + NNODES;
    const int N = NNODES;

    // workspace layout (floats)
    float* ws     = (float*)d_ws;
    float* hbuf   = ws;                         // N*64
    float* agg    = hbuf + (size_t)N * HIDDIM;  // N*64
    float* asrc   = agg  + (size_t)N * HIDDIM;  // N
    float* adst   = asrc + N;                   // N
    float* mbuf   = adst + N;                   // N
    float* sbuf   = mbuf + N;                   // N
    float* stats  = sbuf + N;                   // 2*HIDDIM
    float* pooled = stats + 2 * HIDDIM;         // G*64
    float* counts = pooled + (size_t)NGRAPH * HIDDIM; // G

    const int TB = 256;
    const int gridN    = (N + TB - 1) / TB;
    const int gridNF   = (int)(((long long)N * HIDDIM + TB - 1) / TB);
    const int gridE    = (int)((Etot + TB - 1) / TB);
    const int gridAgg  = (int)((Etot * 32 + TB - 1) / TB);
    const int mtiles   = N / 16;
    const int gridGemm = (mtiles + (TB / 32) - 1) / (TB / 32);

    // ================= Layer 1 =================
    fill4_kernel<<<512, TB, 0, stream>>>((float4*)mbuf, N / 4, NEG_FLT_MAX);
    fill4_kernel<<<512, TB, 0, stream>>>((float4*)sbuf, N / 4, 0.0f);
    fill4_kernel<<<1024, TB, 0, stream>>>((float4*)agg, (long long)N * HIDDIM / 4, 0.0f);

    wmma_gemm_kernel<<<gridGemm, TB, 0, stream>>>(x, W1, hbuf, N, 128);
    alpha_kernel<<<gridN, TB, 0, stream>>>(hbuf, av1s, av1d, asrc, adst, N);

    edge_max_kernel<<<gridE, TB, 0, stream>>>(ei, E, Etot, asrc, adst, mbuf);
    edge_sum_kernel<<<gridE, TB, 0, stream>>>(ei, E, Etot, asrc, adst, mbuf, sbuf);
    edge_agg_kernel<<<gridAgg, TB, 0, stream>>>(ei, E, Etot, asrc, adst, mbuf, sbuf,
                                                hbuf, agg);

    fill4_kernel<<<1, TB, 0, stream>>>((float4*)stats, 2 * HIDDIM / 4, 0.0f);
    bn_stats_kernel<<<512, TB, 0, stream>>>(agg, stats, N);
    bn_apply_kernel<<<gridNF, TB, 0, stream>>>(agg, hbuf, stats, g1, be1, N, 1);

    // ================= Layer 2 =================
    wmma_gemm_kernel<<<gridGemm, TB, 0, stream>>>(hbuf, W2, agg, N, 64);
    alpha_kernel<<<gridN, TB, 0, stream>>>(agg, av2s, av2d, asrc, adst, N);

    fill4_kernel<<<512, TB, 0, stream>>>((float4*)mbuf, N / 4, NEG_FLT_MAX);
    fill4_kernel<<<512, TB, 0, stream>>>((float4*)sbuf, N / 4, 0.0f);
    fill4_kernel<<<1024, TB, 0, stream>>>((float4*)hbuf, (long long)N * HIDDIM / 4, 0.0f);

    edge_max_kernel<<<gridE, TB, 0, stream>>>(ei, E, Etot, asrc, adst, mbuf);
    edge_sum_kernel<<<gridE, TB, 0, stream>>>(ei, E, Etot, asrc, adst, mbuf, sbuf);
    edge_agg_kernel<<<gridAgg, TB, 0, stream>>>(ei, E, Etot, asrc, adst, mbuf, sbuf,
                                                agg, hbuf);

    fill4_kernel<<<1, TB, 0, stream>>>((float4*)stats, 2 * HIDDIM / 4, 0.0f);
    bn_stats_kernel<<<512, TB, 0, stream>>>(hbuf, stats, N);
    bn_apply_kernel<<<gridNF, TB, 0, stream>>>(hbuf, hbuf, stats, g2, be2, N, 0);

    // ================= Pool + FC =================
    fill4_kernel<<<1, TB, 0, stream>>>((float4*)pooled,
                                       ((long long)NGRAPH * HIDDIM + NGRAPH) / 4, 0.0f);
    pool_kernel<<<gridNF, TB, 0, stream>>>(hbuf, batch, pooled, counts, N);
    final_kernel<<<1, 64, 0, stream>>>(pooled, counts, fcW, fcb, outp);
}
